// AdaptiveSpectralRefinement_80788334838232
// MI455X (gfx1250) — compile-verified, matrix operand order checked by reading, not run on hardware
//
#include <hip/hip_runtime.h>
#include <hip/hip_bf16.h>
#include <math.h>

// ---------------------------------------------------------------------------
// AdaptiveSpectralRefinement on MI455X (gfx1250, wave32, WMMA)
//
// irfft(mask * w * rfft(x)) along seq == circulant matmul:
//   x_low = F_low @ X,  F_low[s,t] = k_low[(s-t) mod S]
// Filter (seq axis) and projection (channel axis) commute:
//   PL = (F_low @ X) @ Wl = F_low @ (X @ Wl)
// Gate: sigmoid(PL@Wg1 + PH@Wg2 + bgp), bgp = bg + bl@Wg1 + bh@Wg2.
// Weights are pre-transposed to [N][K] f16 so all WMMA fragments load as
// contiguous 16B chunks (ds_load_b128) -- no LDS gathers in hot loops.
// ---------------------------------------------------------------------------

#define SEQ   96
#define DM    512
#define BATCH 512
#define FREQ  49
#define CUT   16
#define ROWS  (BATCH * SEQ)   // 49152

typedef __attribute__((ext_vector_type(16))) _Float16 v16h;
typedef __attribute__((ext_vector_type(8)))  float    v8f;

union AFrag { v16h v; _Float16 h[16]; };
union CFrag { v8f  v; float    f[8];  };
union H8    { uint4 u; _Float16 h[8]; };

// Fragment load from a row-major tile whose per-lane rows are contiguous:
//  - A operand: rows are M (16x32 tile), row stride ld
//  - B operand stored TRANSPOSED [N][K]: rows are N, same addressing
// lanes 0-15: row=lane,    elems 0-7 -> K 0..7,  8-15 -> K 16..23
// lanes16-31: row=lane-16, elems 0-7 -> K 8..15, 8-15 -> K 24..31
// Chunks are 16B aligned when ld % 8 == 0 -> ds_load_b128 pairs.
__device__ __forceinline__ v16h ldFrag(const _Float16* base, int ld, int lane) {
    AFrag a;
    const _Float16* row = base + (lane & 15) * ld;
    const int kh = (lane >> 4) << 3;   // 0 or 8
#pragma unroll
    for (int i = 0; i < 8; ++i) {
        a.h[i]     = row[kh + i];
        a.h[8 + i] = row[16 + kh + i];
    }
    return a.v;
}

__device__ __forceinline__ v8f wmma_f16(v16h a, v16h b, v8f c) {
    return __builtin_amdgcn_wmma_f32_16x16x32_f16(false, a, false, b,
                                                  (short)0, c, false, false);
}

// ---------------------------------------------------------------------------
// K0: tiled transpose + f32->f16 convert:  src[R][C] f32  ->  dst[C][R] f16
// ---------------------------------------------------------------------------
__global__ void k_transpose_cvt(const float* __restrict__ src,
                                _Float16* __restrict__ dst, int R, int C) {
    __shared__ _Float16 tile[32][33];
    const int tx = threadIdx.x & 31, ty = threadIdx.x >> 5;   // 32 x 8
    const int bx = blockIdx.x * 32;   // C offset
    const int by = blockIdx.y * 32;   // R offset
#pragma unroll
    for (int i = 0; i < 32; i += 8)
        tile[ty + i][tx] = (_Float16)src[(size_t)(by + ty + i) * C + bx + tx];
    __syncthreads();
#pragma unroll
    for (int i = 0; i < 32; i += 8)
        dst[(size_t)(bx + ty + i) * R + by + tx] = tile[tx][ty + i];
}

// ---------------------------------------------------------------------------
// K1: circulant filter matrices (f16) + fused gate bias bgp (f32)
// ---------------------------------------------------------------------------
__global__ void k_prep(const float* __restrict__ fw,
                       const float* __restrict__ bl,
                       const float* __restrict__ bh,
                       const float* __restrict__ Wg,
                       const float* __restrict__ bg,
                       _Float16* __restrict__ Flow,
                       _Float16* __restrict__ Fhigh,
                       float* __restrict__ bgp) {
    __shared__ float kl[SEQ], kh[SEQ];
    const int tid = threadIdx.x;
    if (tid < SEQ) {
        float sl = 0.f, sh = 0.f;
        const float w2pi = 6.28318530717958647692f / (float)SEQ;
#pragma unroll 1
        for (int f = 0; f < FREQ; ++f) {
            const float cf = (f == 0 || f == FREQ - 1) ? 1.f : 2.f;
            const float v  = cf * fw[f] * cosf(w2pi * (float)(f * tid));
            if (f < CUT) sl += v; else sh += v;
        }
        kl[tid] = sl / (float)SEQ;
        kh[tid] = sh / (float)SEQ;
    }
    __syncthreads();
    for (int idx = tid; idx < SEQ * SEQ; idx += blockDim.x) {
        const int s = idx / SEQ, t = idx - s * SEQ;
        int d = s - t; if (d < 0) d += SEQ;
        Flow[idx]  = (_Float16)kl[d];
        Fhigh[idx] = (_Float16)kh[d];
    }
    for (int e = tid; e < DM; e += blockDim.x) {
        float acc = bg[e];
        for (int d = 0; d < DM; ++d)
            acc += bl[d] * Wg[d * DM + e] + bh[d] * Wg[(d + DM) * DM + e];
        bgp[e] = acc;
    }
}

// ---------------------------------------------------------------------------
// K2: per-batch  PL = F_low @ (X @ Wl),  PH = F_high @ (X @ Wh)    (f16 out)
// One WG per batch. All operand tiles fragment-row-major -> b128 LDS traffic.
// LDS: sX[96][520] sWt[64][520] sZt[64][104] sFl/sFh[96][104] sP[96][72]
//      = 233,472 B of the 320 KB WGP budget.
// ---------------------------------------------------------------------------
#define LDX  520
#define LDWT 520
#define LDZT 104
#define LDF  104
#define LDP  72

__global__ void k_proj_filter(const float* __restrict__ x,
                              const _Float16* __restrict__ Wlt,  // [N][K] f16
                              const _Float16* __restrict__ Wht,  // [N][K] f16
                              const _Float16* __restrict__ Flow,
                              const _Float16* __restrict__ Fhigh,
                              _Float16* __restrict__ PL,
                              _Float16* __restrict__ PH) {
    extern __shared__ _Float16 smem[];
    _Float16* sX  = smem;                     // A: [s][k] row-major
    _Float16* sWt = sX + SEQ * LDX;           // B: [n][k] transposed
    _Float16* sZt = sWt + 64 * LDWT;          // Z: [c][t] transposed
    _Float16* sFl = sZt + 64 * LDZT;          // A: [s][t] row-major
    _Float16* sFh = sFl + SEQ * LDF;
    _Float16* sP  = sFh + SEQ * LDF;          // P: [s][c] for coalesced store

    const int tid  = threadIdx.x;
    const int lane = tid & 31;
    const int wave = tid >> 5;                // 8 waves
    const int n    = lane & 15;
    const int mb   = (lane >> 4) << 3;
    const int b    = blockIdx.x;
    const float* Xb = x + (size_t)b * SEQ * DM;

    // stage X: pack 8 f32 -> 8 f16 per thread -> single b128 LDS store
    for (int idx = tid; idx < SEQ * DM / 8; idx += blockDim.x) {
        const int r = idx >> 6, c8 = idx & 63;
        const float4 v0 = *(const float4*)&Xb[r * DM + c8 * 8];
        const float4 v1 = *(const float4*)&Xb[r * DM + c8 * 8 + 4];
        H8 p;
        p.h[0] = (_Float16)v0.x; p.h[1] = (_Float16)v0.y;
        p.h[2] = (_Float16)v0.z; p.h[3] = (_Float16)v0.w;
        p.h[4] = (_Float16)v1.x; p.h[5] = (_Float16)v1.y;
        p.h[6] = (_Float16)v1.z; p.h[7] = (_Float16)v1.w;
        *(uint4*)&sX[r * LDX + c8 * 8] = p.u;
    }
    // stage filter matrices (f16 global, uint4 copies)
    for (int idx = tid; idx < SEQ * SEQ / 8; idx += blockDim.x) {
        const int r = idx / 12, ch = idx - r * 12;
        *(uint4*)&sFl[r * LDF + ch * 8] = *(const uint4*)&Flow[r * SEQ + ch * 8];
        *(uint4*)&sFh[r * LDF + ch * 8] = *(const uint4*)&Fhigh[r * SEQ + ch * 8];
    }
    __syncthreads();

    const int ntw = wave & 3;          // this wave's N-tile (16 cols)
    const int m0w = (wave >> 2) * 3;   // this wave's first M-tile (of 3)

    for (int ct = 0; ct < DM / 64; ++ct) {          // 8 channel tiles of 64
        for (int pass = 0; pass < 2; ++pass) {
            const _Float16* Wt  = pass ? Wht : Wlt;
            const _Float16* sF  = pass ? sFh : sFl;
            _Float16*       dst = (pass ? PH : PL) + (size_t)b * SEQ * DM;

            // stage weight tile [64 n][512 k]: pure uint4 copies
            for (int idx = tid; idx < 64 * DM / 8; idx += blockDim.x) {
                const int r = idx >> 6, ch = idx & 63;
                *(uint4*)&sWt[r * LDWT + ch * 8] =
                    *(const uint4*)&Wt[(size_t)(ct * 64 + r) * DM + ch * 8];
            }
            __syncthreads();

            // Z strip: 3 M-tiles x 1 N-tile per wave; B fragment reused 3x
            {
                CFrag a0, a1, a2;
#pragma unroll
                for (int r = 0; r < 8; ++r) { a0.f[r] = 0.f; a1.f[r] = 0.f; a2.f[r] = 0.f; }
#pragma unroll 4
                for (int kk = 0; kk < DM / 32; ++kk) {
                    const v16h bf = ldFrag(&sWt[(ntw * 16) * LDWT + kk * 32], LDWT, lane);
                    const v16h f0 = ldFrag(&sX[((m0w + 0) * 16) * LDX + kk * 32], LDX, lane);
                    const v16h f1 = ldFrag(&sX[((m0w + 1) * 16) * LDX + kk * 32], LDX, lane);
                    const v16h f2 = ldFrag(&sX[((m0w + 2) * 16) * LDX + kk * 32], LDX, lane);
                    a0.v = wmma_f16(f0, bf, a0.v);
                    a1.v = wmma_f16(f1, bf, a1.v);
                    a2.v = wmma_f16(f2, bf, a2.v);
                }
                // store transposed: lane's n fixed -> contiguous 8 halves (b128)
                CFrag* accs[3] = { &a0, &a1, &a2 };
#pragma unroll
                for (int i = 0; i < 3; ++i) {
                    _Float16* p = &sZt[(ntw * 16 + n) * LDZT + (m0w + i) * 16 + mb];
#pragma unroll
                    for (int r = 0; r < 8; ++r) p[r] = (_Float16)accs[i]->f[r];
                }
            }
            __syncthreads();

            // P strip = F @ Z : both operands contiguous b128 fragment loads
            {
                CFrag a0, a1, a2;
#pragma unroll
                for (int r = 0; r < 8; ++r) { a0.f[r] = 0.f; a1.f[r] = 0.f; a2.f[r] = 0.f; }
#pragma unroll
                for (int kk = 0; kk < 3; ++kk) {        // K = 96 = 3 x 32
                    const v16h bf = ldFrag(&sZt[(ntw * 16) * LDZT + kk * 32], LDZT, lane);
                    const v16h f0 = ldFrag(&sF[((m0w + 0) * 16) * LDF + kk * 32], LDF, lane);
                    const v16h f1 = ldFrag(&sF[((m0w + 1) * 16) * LDF + kk * 32], LDF, lane);
                    const v16h f2 = ldFrag(&sF[((m0w + 2) * 16) * LDF + kk * 32], LDF, lane);
                    a0.v = wmma_f16(f0, bf, a0.v);
                    a1.v = wmma_f16(f1, bf, a1.v);
                    a2.v = wmma_f16(f2, bf, a2.v);
                }
                // scatter into LDS [s][c] tile (cheap), then coalesced store
                CFrag* accs[3] = { &a0, &a1, &a2 };
#pragma unroll
                for (int i = 0; i < 3; ++i)
#pragma unroll
                    for (int r = 0; r < 8; ++r)
                        sP[((m0w + i) * 16 + mb + r) * LDP + ntw * 16 + n] =
                            (_Float16)accs[i]->f[r];
            }
            __syncthreads();

            // coalesced b128 global store of the 96x64 P tile
            for (int i = 0; i < 3; ++i) {
                const int idx = tid + i * 256;          // 768 chunks
                const int r = idx >> 3, ch = idx & 7;
                *(uint4*)&dst[(size_t)r * DM + ct * 64 + ch * 8] =
                    *(const uint4*)&sP[r * LDP + ch * 8];
            }
            __syncthreads();
        }
    }
}

// ---------------------------------------------------------------------------
// K3: gate = sigmoid(PL@Wg1 + PH@Wg2 + bgp)  — [49152 x 1024] @ [1024 x 512]
// 64x64 block tile, K-step 32; Wgt pre-transposed [512 n][1024 k] f16 so both
// staging and fragment loads are contiguous. Output staged via LDS.
// ---------------------------------------------------------------------------
__global__ void k_gate(const _Float16* __restrict__ PL,
                       const _Float16* __restrict__ PH,
                       const _Float16* __restrict__ Wgt,   // [512][1024] f16
                       const float* __restrict__ bgp,
                       _Float16* __restrict__ gate) {
    __shared__ _Float16 sA[64 * 40];     // [m][k] row-major, ld 40
    __shared__ _Float16 sBt[64 * 40];    // [n][k] transposed, ld 40
    __shared__ _Float16 sOut[64 * 72];   // [m][n] for coalesced store

    const int tid  = threadIdx.x;
    const int lane = tid & 31;
    const int wave = tid >> 5;
    const int n    = lane & 15;
    const int mb   = (lane >> 4) << 3;
    const int row0 = blockIdx.x * 64;
    const int col0 = blockIdx.y * 64;

    const int ntw = wave & 3;          // N-tile
    const int m0w = (wave >> 2) * 2;   // first of 2 M-tiles

    CFrag acc[2];
#pragma unroll
    for (int t = 0; t < 2; ++t)
#pragma unroll
        for (int r = 0; r < 8; ++r) acc[t].f[r] = 0.f;

    for (int ks = 0; ks < 2 * DM / 32; ++ks) {      // 32 K-steps over 1024
        const int k0 = ks * 32;
        // A tile: 64 rows x 32 k == 256 16B chunks, one per thread
        {
            const int r  = tid >> 2, ch = tid & 3;
            const int kg = k0 + ch * 8;
            const size_t row = (size_t)(row0 + r);
            const _Float16* src = (kg < DM) ? &PL[row * DM + kg]
                                            : &PH[row * DM + (kg - DM)];
            *(uint4*)&sA[r * 40 + ch * 8] = *(const uint4*)src;
        }
        // B tile: 64 n-rows x 32 k == 256 16B chunks, one per thread
        {
            const int r = tid >> 2, ch = tid & 3;
            *(uint4*)&sBt[r * 40 + ch * 8] =
                *(const uint4*)&Wgt[(size_t)(col0 + r) * (2 * DM) + k0 + ch * 8];
        }
        __syncthreads();
        const v16h bf = ldFrag(&sBt[(ntw * 16) * 40], 40, lane);
#pragma unroll
        for (int t = 0; t < 2; ++t) {
            const v16h af = ldFrag(&sA[((m0w + t) * 16) * 40], 40, lane);
            acc[t].v = wmma_f16(af, bf, acc[t].v);
        }
        __syncthreads();
    }

    // epilogue: bias + sigmoid -> LDS tile -> coalesced b128 global stores
    const float bb = bgp[col0 + ntw * 16 + n];
#pragma unroll
    for (int t = 0; t < 2; ++t)
#pragma unroll
        for (int r = 0; r < 8; ++r) {
            const float v = acc[t].f[r] + bb;
            const float g = 1.f / (1.f + expf(-v));
            sOut[((m0w + t) * 16 + mb + r) * 72 + ntw * 16 + n] = (_Float16)g;
        }
    __syncthreads();
#pragma unroll
    for (int i = 0; i < 2; ++i) {
        const int idx = tid + i * 256;          // 512 chunks of 8 halves
        const int r = idx >> 3, ch = idx & 7;
        *(uint4*)&gate[(size_t)(row0 + r) * DM + col0 + ch * 8] =
            *(const uint4*)&sOut[r * 72 + ch * 8];
    }
}

// ---------------------------------------------------------------------------
// K4: out = LayerNorm(x + gate*(PL+bl) + (1-gate)*(PH+bh)) * gamma + beta
// 4 rows per block; 64 threads x 8 contiguous channels -> fully vectorized.
// ---------------------------------------------------------------------------
__global__ void k_combine_ln(const float* __restrict__ x,
                             const _Float16* __restrict__ PL,
                             const _Float16* __restrict__ PH,
                             const _Float16* __restrict__ gate,
                             const float* __restrict__ bl,
                             const float* __restrict__ bh,
                             const float* __restrict__ gamma,
                             const float* __restrict__ beta,
                             float* __restrict__ out) {
    const int tid = threadIdx.x;
    const int rl  = tid >> 6;                 // row within block (0..3)
    const int t64 = tid & 63;
    const size_t row = (size_t)blockIdx.x * 4 + rl;
    const int c0 = t64 * 8;
    const size_t base = row * DM + c0;

    H8 plv, phv, gv;
    plv.u = *(const uint4*)&PL[base];
    phv.u = *(const uint4*)&PH[base];
    gv.u  = *(const uint4*)&gate[base];
    float xs[8], bls[8], bhs[8];
    *(float4*)&xs[0]  = *(const float4*)&x[base];
    *(float4*)&xs[4]  = *(const float4*)&x[base + 4];
    *(float4*)&bls[0] = *(const float4*)&bl[c0];
    *(float4*)&bls[4] = *(const float4*)&bl[c0 + 4];
    *(float4*)&bhs[0] = *(const float4*)&bh[c0];
    *(float4*)&bhs[4] = *(const float4*)&bh[c0 + 4];

    float y[8], s1 = 0.f, s2 = 0.f;
#pragma unroll
    for (int j = 0; j < 8; ++j) {
        const float pl = (float)plv.h[j] + bls[j];
        const float ph = (float)phv.h[j] + bhs[j];
        const float g  = (float)gv.h[j];
        y[j] = xs[j] + g * pl + (1.f - g) * ph;
        s1 += y[j]; s2 += y[j] * y[j];
    }
#pragma unroll
    for (int off = 16; off > 0; off >>= 1) {
        s1 += __shfl_down(s1, off, 32);
        s2 += __shfl_down(s2, off, 32);
    }
    __shared__ float r1[8], r2[8];
    const int w = tid >> 5;
    if ((tid & 31) == 0) { r1[w] = s1; r2[w] = s2; }
    __syncthreads();
    const float sum1 = r1[rl * 2] + r1[rl * 2 + 1];
    const float sum2 = r2[rl * 2] + r2[rl * 2 + 1];
    const float mean = sum1 * (1.f / DM);
    const float var  = sum2 * (1.f / DM) - mean * mean;
    const float inv  = rsqrtf(var + 1e-5f);

    float gm[8], bt[8], o[8];
    *(float4*)&gm[0] = *(const float4*)&gamma[c0];
    *(float4*)&gm[4] = *(const float4*)&gamma[c0 + 4];
    *(float4*)&bt[0] = *(const float4*)&beta[c0];
    *(float4*)&bt[4] = *(const float4*)&beta[c0 + 4];
#pragma unroll
    for (int j = 0; j < 8; ++j) o[j] = (y[j] - mean) * inv * gm[j] + bt[j];
    *(float4*)&out[base]     = *(const float4*)&o[0];
    *(float4*)&out[base + 4] = *(const float4*)&o[4];
}

// ---------------------------------------------------------------------------
extern "C" void kernel_launch(void* const* d_in, const int* in_sizes, int n_in,
                              void* d_out, int out_size, void* d_ws, size_t ws_size,
                              hipStream_t stream) {
    const float* x     = (const float*)d_in[0];
    const float* fw    = (const float*)d_in[1];
    const float* Wl    = (const float*)d_in[2];
    const float* bl    = (const float*)d_in[3];
    const float* Wh    = (const float*)d_in[4];
    const float* bh    = (const float*)d_in[5];
    const float* Wg    = (const float*)d_in[6];
    const float* bg    = (const float*)d_in[7];
    const float* gamma = (const float*)d_in[8];
    const float* beta  = (const float*)d_in[9];
    float* out = (float*)d_out;

    const size_t nelem = (size_t)ROWS * DM;          // 25,165,824
    _Float16* PL    = (_Float16*)d_ws;
    _Float16* PH    = PL + nelem;
    _Float16* gate  = PH + nelem;
    _Float16* Flow  = gate + nelem;
    _Float16* Fhigh = Flow + SEQ * SEQ;
    _Float16* Wlt   = Fhigh + SEQ * SEQ;             // [512][512]  f16
    _Float16* Wht   = Wlt + (size_t)DM * DM;
    _Float16* Wgt   = Wht + (size_t)DM * DM;         // [512][1024] f16
    float*    bgp   = (float*)(Wgt + (size_t)DM * 2 * DM);

    k_prep<<<1, 256, 0, stream>>>(fw, bl, bh, Wg, bg, Flow, Fhigh, bgp);

    dim3 tgrid(DM / 32, DM / 32);
    k_transpose_cvt<<<tgrid, 256, 0, stream>>>(Wl, Wlt, DM, DM);
    k_transpose_cvt<<<tgrid, 256, 0, stream>>>(Wh, Wht, DM, DM);
    dim3 tgridg(DM / 32, 2 * DM / 32);
    k_transpose_cvt<<<tgridg, 256, 0, stream>>>(Wg, Wgt, 2 * DM, DM);

    const size_t smem2 =
        (size_t)(SEQ * LDX + 64 * LDWT + 64 * LDZT + 2 * SEQ * LDF + SEQ * LDP) *
        sizeof(_Float16);
    k_proj_filter<<<BATCH, 256, smem2, stream>>>(x, Wlt, Wht, Flow, Fhigh, PL, PH);

    dim3 g3(ROWS / 64, DM / 64);
    k_gate<<<g3, 256, 0, stream>>>(PL, PH, Wgt, bgp, gate);

    k_combine_ln<<<ROWS / 4, 256, 0, stream>>>(x, PL, PH, gate, bl, bh,
                                               gamma, beta, out);
}
